// ConvLSTMHead_80745385165044
// MI455X (gfx1250) — compile-verified, matrix operand order checked by reading, not run on hardware
//
#include <hip/hip_runtime.h>
#include <hip/hip_bf16.h>

typedef __attribute__((ext_vector_type(16))) __bf16 v16bf;
typedef __attribute__((ext_vector_type(8)))  __bf16 v8bf;
typedef __attribute__((ext_vector_type(8)))  float  v8f;

#define Bsz  4
#define Ssz  32
#define Csz  128
#define HID  128
#define NPOS 1024            // 32*32
#define Mtot 4096            // B * NPOS
#define Ngate 512            // 4*HID
#define Kdim 1152            // 9 * 128 (per conv half)

union ABfrag { v16bf v; v8bf h[2]; };

// ---------- data prep ----------
// x: [B,S,C,H,W] fp32 -> xb: [B,S,pos,C] bf16 (NHWC so channel strips are contiguous)
__global__ void convert_x_kernel(const float* __restrict__ x, __bf16* __restrict__ xb) {
  long i = (long)blockIdx.x * blockDim.x + threadIdx.x;
  const long total = (long)Bsz * Ssz * Csz * NPOS;
  if (i >= total) return;
  int c   = (int)(i & (Csz - 1));
  int pos = (int)((i >> 7) & (NPOS - 1));
  long bt = i >> 17;
  xb[i] = (__bf16)x[(bt * Csz + c) * NPOS + pos];
}

// conv_w [512][256][3][3] fp32 -> Wx/Wh bf16 [n][k], k = r*128 + c (tap-major, channel-minor)
__global__ void prep_w_kernel(const float* __restrict__ w,
                              __bf16* __restrict__ wx, __bf16* __restrict__ wh) {
  int i = blockIdx.x * blockDim.x + threadIdx.x;
  if (i >= Ngate * Kdim) return;
  int n = i / Kdim, k = i - n * Kdim;
  int r = k >> 7, c = k & 127;
  wx[i] = (__bf16)w[((long)n * 256 + c) * 9 + r];
  wh[i] = (__bf16)w[((long)n * 256 + 128 + c) * 9 + r];
}

__global__ void init_state_kernel(const float* __restrict__ ih, const float* __restrict__ ic,
                                  __bf16* __restrict__ hb, float* __restrict__ cs) {
  int i = blockIdx.x * blockDim.x + threadIdx.x;
  if (i >= Mtot * HID) return;
  int ch = i & (HID - 1);
  hb[i] = (__bf16)ih[ch];
  cs[i] = ic[ch];
}

// ---------- implicit-GEMM 3x3 conv via bf16 WMMA, f32 accumulate ----------
// Block = 128 threads (4 wave32). Each wave computes an (MI*16)x64 fp32 C tile;
// block tile = (MI*64)x64. Double-buffered LDS staging overlaps global loads
// with WMMA work (one barrier per K-chunk).
template<bool HAS_ADD, int MI>
__global__ __launch_bounds__(128) void gemm_bf16_kernel(
    const __bf16* __restrict__ A, const __bf16* __restrict__ Bw,
    float* __restrict__ Cout, const float* __restrict__ Cadd,
    long a_t_stride, long a_b_stride, long c_t_stride)
{
  constexpr int MROWS = 64 * MI;
  __shared__ __align__(16) __bf16 lsA[2][MROWS][40]; // rows padded to 80B: conflict-free b128
  __shared__ __align__(16) __bf16 lsB[2][64][40];
  const int tid = threadIdx.x;
  const int m0 = blockIdx.x * MROWS;
  const int n0 = blockIdx.y * 64;
  const int t  = blockIdx.z;

  const int wave = tid >> 5;
  const int lane = tid & 31;
  const int lm = lane & 15;
  const int lh = lane >> 4;        // K-half select per documented 16-bit layouts
  const int kh = lh * 8;

  v8f acc[MI][4];
  #pragma unroll
  for (int i = 0; i < MI; ++i)
    #pragma unroll
    for (int j = 0; j < 4; ++j) acc[i][j] = (v8f){};

  auto stage = [&](int k0, int buf) {
    const int r  = k0 >> 7;        // 3x3 tap index (chunks never straddle taps: 128%32==0)
    const int c0 = k0 & 127;
    const int dy = r / 3 - 1;
    const int dx = r % 3 - 1;
    // A tile (implicit im2col with zero padding): MROWS x 32 bf16, 16B granules
    #pragma unroll
    for (int u = tid; u < MROWS * 4; u += 128) {
      int row = u >> 2, seg = u & 3;
      int m = m0 + row;
      int b = m >> 10, pos = m & (NPOS - 1);
      int y = pos >> 5, xx = pos & 31;
      int yy = y + dy, x2 = xx + dx;
      v8bf val = {};
      if ((unsigned)yy < 32u && (unsigned)x2 < 32u)
        val = *(const v8bf*)(A + (long)t * a_t_stride + (long)b * a_b_stride
                             + (long)((yy << 5) + x2) * Csz + c0 + seg * 8);
      *(v8bf*)&lsA[buf][row][seg * 8] = val;
    }
    // B weight tile: 64 n-rows x 32 k
    #pragma unroll
    for (int u = tid; u < 256; u += 128) {
      int row = u >> 2, seg = u & 3;
      *(v8bf*)&lsB[buf][row][seg * 8] =
          *(const v8bf*)(Bw + (long)(n0 + row) * Kdim + k0 + seg * 8);
    }
  };

  stage(0, 0);
  __syncthreads();
  constexpr int NK = Kdim / 32;
  for (int kc = 0; kc < NK; ++kc) {
    const int buf = kc & 1;
    if (kc + 1 < NK) stage((kc + 1) * 32, buf ^ 1);  // overlap with WMMA below

    ABfrag af[MI];
    #pragma unroll
    for (int i = 0; i < MI; ++i) {
      const int ar = wave * 16 * MI + i * 16 + lm;
      af[i].h[0] = *(const v8bf*)&lsA[buf][ar][kh];
      af[i].h[1] = *(const v8bf*)&lsA[buf][ar][16 + kh];
    }
    #pragma unroll
    for (int j = 0; j < 4; ++j) {
      ABfrag bf_;
      bf_.h[0] = *(const v8bf*)&lsB[buf][j * 16 + lm][kh];
      bf_.h[1] = *(const v8bf*)&lsB[buf][j * 16 + lm][16 + kh];
      #pragma unroll
      for (int i = 0; i < MI; ++i)
        acc[i][j] = __builtin_amdgcn_wmma_f32_16x16x32_bf16(
            false, af[i].v, false, bf_.v, (short)0, acc[i][j], false, false);
    }
    __syncthreads();
  }

  // C layout: lanes 0-15 -> M=e, lanes 16-31 -> M=e+8 (VGPR e), N = lane%16
  const long cbase = (long)t * c_t_stride;
  #pragma unroll
  for (int i = 0; i < MI; ++i) {
    const int mbase = m0 + wave * 16 * MI + i * 16 + lh * 8;
    #pragma unroll
    for (int j = 0; j < 4; ++j) {
      const int col = n0 + j * 16 + lm;
      #pragma unroll
      for (int e = 0; e < 8; ++e) {
        long idx = (long)(mbase + e) * Ngate + col;
        float v = acc[i][j][e];
        if (HAS_ADD) v += Cadd[idx];
        Cout[cbase + idx] = v;
      }
    }
  }
}

// ---------- pointwise LSTM cell update ----------
__global__ void lstm_update_kernel(const float* __restrict__ gates, const float* __restrict__ bias,
                                   float* __restrict__ cs, __bf16* __restrict__ hb) {
  int m = blockIdx.x;       // 0..4095 (b*1024+pos)
  int ch = threadIdx.x;     // 0..127
  long base = (long)m * Ngate + ch;
  float gi = gates[base +   0] + bias[ch];
  float gf = gates[base + 128] + bias[ch + 128];
  float go = gates[base + 256] + bias[ch + 256];
  float gg = gates[base + 384] + bias[ch + 384];
  float i = 1.f / (1.f + __expf(-gi));
  float f = 1.f / (1.f + __expf(-gf));
  float o = 1.f / (1.f + __expf(-go));
  float g = tanhf(gg);
  long si = (long)m * HID + ch;
  float cn = f * cs[si] + i * g;
  cs[si] = cn;
  hb[si] = (__bf16)(o * tanhf(cn));
}

// ---------- deterministic spatial mean: feat[b,t,ch] ----------
__global__ void feat_reduce_kernel(const __bf16* __restrict__ hb, float* __restrict__ feat, int t) {
  int b = blockIdx.x, ch = threadIdx.x;
  const __bf16* p = hb + (long)b * NPOS * HID + ch;
  float s = 0.f;
  for (int pos = 0; pos < NPOS; ++pos) s += (float)p[(long)pos * HID];
  feat[((long)b * Ssz + t) * HID + ch] = s * (1.0f / NPOS);
}

// ---------- FC + ReLU + two scalar heads ----------
__global__ void head_kernel(const float* __restrict__ feat,
                            const float* __restrict__ fc_w, const float* __restrict__ fc_b,
                            const float* __restrict__ fco_w, const float* __restrict__ fco_b,
                            const float* __restrict__ fca_w, const float* __restrict__ fca_b,
                            float* __restrict__ out) {
  __shared__ float sf[HID];
  __shared__ float so[Csz];
  __shared__ float sa[Csz];
  int row = blockIdx.x;   // b*S+t
  int c = threadIdx.x;    // 0..127
  sf[c] = feat[(long)row * HID + c];
  __syncthreads();
  float acc = fc_b[c];
  #pragma unroll 8
  for (int h = 0; h < HID; ++h) acc = fmaf(sf[h], fc_w[(long)c * HID + h], acc);
  float f2 = fmaxf(acc, 0.f);
  so[c] = f2 * fco_w[c];
  sa[c] = f2 * fca_w[c];
  __syncthreads();
  for (int s = 64; s > 0; s >>= 1) {
    if (c < s) { so[c] += so[c + s]; sa[c] += sa[c + s]; }
    __syncthreads();
  }
  if (c == 0) {
    out[row]             = so[0] + fco_b[0];
    out[Bsz * Ssz + row] = sa[0] + fca_b[0];
  }
}

extern "C" void kernel_launch(void* const* d_in, const int* in_sizes, int n_in,
                              void* d_out, int out_size, void* d_ws, size_t ws_size,
                              hipStream_t stream) {
  const float* x      = (const float*)d_in[0];
  const float* conv_w = (const float*)d_in[1];
  const float* conv_b = (const float*)d_in[2];
  const float* init_h = (const float*)d_in[3];
  const float* init_c = (const float*)d_in[4];
  const float* fc_w   = (const float*)d_in[5];
  const float* fc_b   = (const float*)d_in[6];
  const float* fco_w  = (const float*)d_in[7];
  const float* fco_b  = (const float*)d_in[8];
  const float* fca_w  = (const float*)d_in[9];
  const float* fca_b  = (const float*)d_in[10];
  float* out = (float*)d_out;
  (void)in_sizes; (void)n_in; (void)out_size; (void)ws_size;

  char* ws = (char*)d_ws;
  size_t off = 0;
  auto alloc = [&](size_t bytes) -> void* {
    void* p = ws + off;
    off = (off + bytes + 255) & ~(size_t)255;
    return p;
  };
  float*  Gx    = (float*) alloc((size_t)Ssz * Mtot * Ngate * 4);     // 256 MiB
  float*  gates = (float*) alloc((size_t)Mtot * Ngate * 4);           // 8 MiB
  __bf16* xb    = (__bf16*)alloc((size_t)Bsz * Ssz * NPOS * Csz * 2); // 32 MiB
  __bf16* Wx    = (__bf16*)alloc((size_t)Ngate * Kdim * 2);
  __bf16* Wh    = (__bf16*)alloc((size_t)Ngate * Kdim * 2);
  __bf16* hb    = (__bf16*)alloc((size_t)Mtot * HID * 2);
  float*  cs    = (float*) alloc((size_t)Mtot * HID * 4);
  float*  feat  = (float*) alloc((size_t)Bsz * Ssz * HID * 4);

  convert_x_kernel<<<(Bsz * Ssz * Csz * NPOS + 255) / 256, 256, 0, stream>>>(x, xb);
  prep_w_kernel<<<(Ngate * Kdim + 255) / 256, 256, 0, stream>>>(conv_w, Wx, Wh);
  init_state_kernel<<<(Mtot * HID + 255) / 256, 256, 0, stream>>>(init_h, init_c, hb, cs);

  // Parallel half: Gx[t] = conv_x(x_t) for every timestep at once (grid.z = S).
  // Throughput-bound -> wide 128x64 block tiles (8 WMMAs per chunk per wave).
  gemm_bf16_kernel<false, 2><<<dim3(Mtot / 128, Ngate / 64, Ssz), 128, 0, stream>>>(
      xb, Wx, Gx, nullptr,
      (long)NPOS * Csz, (long)Ssz * NPOS * Csz, (long)Mtot * Ngate);

  // Sequential half: gates = Gx[t] + conv_h(h); latency-sensitive -> 64x64 tiles
  // (512 blocks/step for occupancy), fused Gx add in the epilogue.
  for (int t = 0; t < Ssz; ++t) {
    gemm_bf16_kernel<true, 1><<<dim3(Mtot / 64, Ngate / 64, 1), 128, 0, stream>>>(
        hb, Wh, gates, Gx + (size_t)t * Mtot * Ngate,
        0L, (long)NPOS * Csz, 0L);
    lstm_update_kernel<<<Mtot, HID, 0, stream>>>(gates, conv_b, cs, hb);
    feat_reduce_kernel<<<Bsz, HID, 0, stream>>>(hb, feat, t);
  }

  head_kernel<<<Bsz * Ssz, Csz, 0, stream>>>(feat, fc_w, fc_b, fco_w, fco_b, fca_w, fca_b, out);
}
